// Direction_Attention_layer_12283606468238
// MI455X (gfx1250) — matrix operand order repaired and checked by reading, compiled, not run on hardware
//
#include <hip/hip_runtime.h>
#include <math.h>

// ---------------------------------------------------------------------------
// Direction_Attention for MI455X (gfx1250): two fused bf16x3 WMMA GEMMs with
// async (ASYNCcnt) double-buffered LDS staging of the B operand.
//   uit = tanh(x@W + b)            (gemm<0>, stores uit as bf16 hi/lo planes)
//   a   = exp(uit@U)               (gemm<1>, masked-reduced on the fly)
//   out = [xsum*(fw+eps), xsum*(bw+eps)]
// ---------------------------------------------------------------------------

typedef __attribute__((ext_vector_type(16))) __bf16       v16bf;
typedef __attribute__((ext_vector_type(8)))  float        v8f;
typedef __attribute__((ext_vector_type(4)))  float        f32x4;

#define NBATCH 64
#define SEQ    512
#define DIM    512
#define EPSV   1e-7f

union V16U { v16bf v; f32x4 q[2]; };

__device__ __forceinline__ v8f wmma_bf16(v16bf a, v16bf b, v8f c) {
  // D = A(16x32 bf16) * B(32x16 bf16) + C(16x16 f32)
  return __builtin_amdgcn_wmma_f32_16x16x32_bf16(false, a, false, b,
                                                 (short)0, c, false, false);
}

__device__ __forceinline__ void wait_asynccnt0() {
  asm volatile("s_wait_asynccnt 0" ::: "memory");
}
__device__ __forceinline__ void wait_dscnt0() {
  asm volatile("s_wait_dscnt 0" ::: "memory");
}

// One b128 async copy: global (bf16) -> LDS, no VGPR round trip, ASYNCcnt.
__device__ __forceinline__ void async_copy_b128(const __bf16* gsrc, __bf16* ldst) {
  // Generic shared-pointer low 32 bits == LDS address (aperture rule).
  unsigned int loff = (unsigned int)(unsigned long long)(uintptr_t)ldst;
  unsigned long long gaddr = (unsigned long long)(uintptr_t)gsrc;
  asm volatile("global_load_async_to_lds_b128 %0, %1, off"
               :: "v"(loff), "v"(gaddr) : "memory");
}

// ---------------------------------------------------------------------------
// Kernel 0: W,U fp32 -> transposed bf16 hi/lo planes ([n][k] so B-fragments
// are contiguous); also zero the fw/bw reduction buffers.
// ---------------------------------------------------------------------------
__global__ void prep_kernel(const float* __restrict__ W, const float* __restrict__ U,
                            __bf16* __restrict__ WhiT, __bf16* __restrict__ WloT,
                            __bf16* __restrict__ UhiT, __bf16* __restrict__ UloT,
                            float* __restrict__ fwsum, float* __restrict__ bwsum) {
  int idx = blockIdx.x * 256 + threadIdx.x;   // 0 .. 512*512-1
  int k = idx >> 9, n = idx & 511;
  float wv = W[idx];
  __bf16 wh = (__bf16)wv;
  WhiT[n * DIM + k] = wh;
  WloT[n * DIM + k] = (__bf16)(wv - (float)wh);
  float uv = U[idx];
  __bf16 uh = (__bf16)uv;
  UhiT[n * DIM + k] = uh;
  UloT[n * DIM + k] = (__bf16)(uv - (float)uh);
  if (idx < NBATCH * DIM) { fwsum[idx] = 0.f; bwsum[idx] = 0.f; }
}

// ---------------------------------------------------------------------------
// Fused GEMM. Workgroup = (batch b, 128-row block, 128-col block), 8 waves.
// Wave w owns row-tile w (16 rows) x 8 col-tiles. K=512 in 16 steps of 32.
// B tile (128n x 32k, hi+lo bf16 planes) is staged via async-to-LDS with
// double buffering: tile ks+1 streams in while tile ks feeds the WMMAs.
// ---------------------------------------------------------------------------
template <int MODE>
__global__ void __launch_bounds__(256)
gemm_kernel(const float* __restrict__ X,
            const __bf16* __restrict__ AHi, const __bf16* __restrict__ ALo,
            const __bf16* __restrict__ BThi, const __bf16* __restrict__ BTlo,
            const float* __restrict__ bias,
            __bf16* __restrict__ OutHi, __bf16* __restrict__ OutLo,
            float* __restrict__ fwsum, float* __restrict__ bwsum) {
  // pitch 40 halves (80B): 16B-aligned rows, conflict-free fragment reads
  __shared__ __bf16 ldsBhi[2][128][40];
  __shared__ __bf16 ldsBlo[2][128][40];
  __shared__ float  ldsFW[128];
  __shared__ float  ldsBW[128];

  const int t    = threadIdx.x;
  const int w    = t >> 5;        // wave 0..7 = row tile
  const int lane = t & 31;
  const int nl   = lane & 15;     // N within tile / A-row within tile
  const int part = lane >> 4;     // K-half selector (A/B), M-half selector (C)

  const int bid  = blockIdx.x;    // 64 * 4 * 4 = 1024 blocks
  const int bb   = bid >> 4;
  const int sblk = (bid >> 2) & 3;
  const int nblk = bid & 3;
  const int s0   = sblk * 128;
  const int n0   = nblk * 128;

  if (MODE == 1 && t < 128) { ldsFW[t] = 0.f; ldsBW[t] = 0.f; }

  v8f acc[8];
  const v8f vzero = {0.f, 0.f, 0.f, 0.f, 0.f, 0.f, 0.f, 0.f};
#pragma unroll
  for (int i = 0; i < 8; ++i) acc[i] = vzero;

  const int ma = s0 + w * 16 + nl;          // A-fragment row for this lane

  // thread t copies 2 b128 octs per plane per k-step
  const int row0 = t >> 2;                  // 0..63
  const int oc   = (t & 3) * 8;             // k offset within tile (halves)

  // async staging of one 128x32 hi+lo tile into buffer `buf`
  auto stage = [&](int buf, int k0) {
#pragma unroll
    for (int i = 0; i < 2; ++i) {
      int row = row0 + i * 64;
      size_t g = (size_t)(n0 + row) * DIM + (size_t)(k0 + oc);
      async_copy_b128(BThi + g, &ldsBhi[buf][row][oc]);
      async_copy_b128(BTlo + g, &ldsBlo[buf][row][oc]);
    }
  };

  stage(0, 0);   // prologue: tile 0

  for (int ks = 0; ks < 16; ++ks) {
    const int k0  = ks * 32;
    const int cur = ks & 1;

    wait_asynccnt0();      // my async copies for tile ks have landed
    __syncthreads();       // ... and everyone else's

    if (ks < 15) stage(cur ^ 1, k0 + 32);   // overlap tile ks+1 with compute

    // --- A fragments (WMMA 16x32 layout: elems 0..7 = K part*8..+7,
    //     elems 8..15 = K 16+part*8..+7) ---
    V16U a_hi, a_lo;
    if (MODE == 0) {
      const float* xr = X + ((size_t)bb * SEQ + ma) * DIM + (size_t)(k0 + part * 8);
      f32x4 f0 = *(const f32x4*)(xr);
      f32x4 f1 = *(const f32x4*)(xr + 4);
      f32x4 f2 = *(const f32x4*)(xr + 16);
      f32x4 f3 = *(const f32x4*)(xr + 20);
      float fl[16];
#pragma unroll
      for (int e = 0; e < 4; ++e) {
        fl[e] = f0[e]; fl[4 + e] = f1[e]; fl[8 + e] = f2[e]; fl[12 + e] = f3[e];
      }
#pragma unroll
      for (int e = 0; e < 16; ++e) {
        __bf16 h = (__bf16)fl[e];
        a_hi.v[e] = h;
        a_lo.v[e] = (__bf16)(fl[e] - (float)h);
      }
    } else {
      size_t g = ((size_t)bb * SEQ + ma) * DIM + (size_t)(k0 + part * 8);
      const __bf16* ah = AHi + g;
      const __bf16* al = ALo + g;
      a_hi.q[0] = *(const f32x4*)(ah);
      a_hi.q[1] = *(const f32x4*)(ah + 16);
      a_lo.q[0] = *(const f32x4*)(al);
      a_lo.q[1] = *(const f32x4*)(al + 16);
    }

    // --- 8 col-tiles x bf16x3 WMMA ---
#pragma unroll
    for (int ct = 0; ct < 8; ++ct) {
      const __bf16* pb = &ldsBhi[cur][ct * 16 + nl][part * 8];
      const __bf16* pl = &ldsBlo[cur][ct * 16 + nl][part * 8];
      V16U bh, bl;
      bh.q[0] = *(const f32x4*)(pb); bh.q[1] = *(const f32x4*)(pb + 16);
      bl.q[0] = *(const f32x4*)(pl); bl.q[1] = *(const f32x4*)(pl + 16);
      acc[ct] = wmma_bf16(a_hi.v, bh.v, acc[ct]);
      acc[ct] = wmma_bf16(a_hi.v, bl.v, acc[ct]);
      acc[ct] = wmma_bf16(a_lo.v, bh.v, acc[ct]);
    }

    wait_dscnt0();         // my B-fragment reads of buf `cur` retired
    __syncthreads();       // all waves done reading -> safe to refill at ks+2
  }

  // --- epilogue; C layout: M = part*8 + v, N = nl ---
  if (MODE == 0) {
#pragma unroll
    for (int ct = 0; ct < 8; ++ct) {
      int n = n0 + ct * 16 + nl;
      float bv = bias[n];
#pragma unroll
      for (int v = 0; v < 8; ++v) {
        int m = s0 + w * 16 + part * 8 + v;
        float val = tanhf(acc[ct][v] + bv);
        __bf16 h = (__bf16)val;
        size_t o = ((size_t)bb * SEQ + m) * DIM + n;
        OutHi[o] = h;
        OutLo[o] = (__bf16)(val - (float)h);
      }
    }
  } else {
#pragma unroll
    for (int ct = 0; ct < 8; ++ct) {
      int d = n0 + ct * 16 + nl;
      float fw = 0.f, bw = 0.f;
#pragma unroll
      for (int v = 0; v < 8; ++v) {
        int s = s0 + w * 16 + part * 8 + v;
        float av = expf(acc[ct][v]);
        if (s > d) fw += av;
        else if (s < d) bw += av;
      }
      fw += __shfl_xor(fw, 16, 32);   // combine the two M-halves (same d)
      bw += __shfl_xor(bw, 16, 32);
      if (part == 0) {
        atomicAdd(&ldsFW[ct * 16 + nl], fw);
        atomicAdd(&ldsBW[ct * 16 + nl], bw);
      }
    }
    __syncthreads();
    if (t < 128) {
      atomicAdd(&fwsum[(size_t)bb * DIM + n0 + t], ldsFW[t]);
      atomicAdd(&bwsum[(size_t)bb * DIM + n0 + t], ldsBW[t]);
    }
  }
}

// ---------------------------------------------------------------------------
// Kernel 3: xsum[b,d] = sum_s x[b,s,d]; out = [xsum*(fw+eps), xsum*(bw+eps)]
// ---------------------------------------------------------------------------
__global__ void finalize_kernel(const float* __restrict__ X,
                                const float* __restrict__ fwsum,
                                const float* __restrict__ bwsum,
                                float* __restrict__ out) {
  int t = blockIdx.x * 256 + threadIdx.x;   // 0..32767
  int bb = t >> 9, d = t & 511;
  const float* xp = X + ((size_t)bb * SEQ) * DIM + d;
  float s = 0.f;
#pragma unroll 8
  for (int i = 0; i < SEQ; ++i) s += xp[(size_t)i * DIM];
  out[(size_t)bb * (2 * DIM) + d]       = s * (fwsum[t] + EPSV);
  out[(size_t)bb * (2 * DIM) + DIM + d] = s * (bwsum[t] + EPSV);
}

// ---------------------------------------------------------------------------
extern "C" void kernel_launch(void* const* d_in, const int* in_sizes, int n_in,
                              void* d_out, int out_size, void* d_ws, size_t ws_size,
                              hipStream_t stream) {
  (void)in_sizes; (void)n_in; (void)out_size; (void)ws_size;
  const float* x    = (const float*)d_in[0];
  const float* W    = (const float*)d_in[1];
  const float* U    = (const float*)d_in[2];
  const float* bias = (const float*)d_in[3];
  float* out = (float*)d_out;

  // workspace layout (bf16 elems unless noted); total ~66.3 MB
  __bf16* WhiT  = (__bf16*)d_ws;
  __bf16* WloT  = WhiT + 512 * 512;
  __bf16* UhiT  = WloT + 512 * 512;
  __bf16* UloT  = UhiT + 512 * 512;
  __bf16* UIThi = UloT + 512 * 512;                       // 64*512*512
  __bf16* UITlo = UIThi + (size_t)NBATCH * SEQ * DIM;
  float*  fwsum = (float*)(UITlo + (size_t)NBATCH * SEQ * DIM);
  float*  bwsum = fwsum + NBATCH * DIM;

  prep_kernel<<<1024, 256, 0, stream>>>(W, U, WhiT, WloT, UhiT, UloT, fwsum, bwsum);

  gemm_kernel<0><<<1024, 256, 0, stream>>>(x, nullptr, nullptr, WhiT, WloT,
                                           bias, UIThi, UITlo, nullptr, nullptr);

  gemm_kernel<1><<<1024, 256, 0, stream>>>(nullptr, UIThi, UITlo, UhiT, UloT,
                                           nullptr, nullptr, nullptr, fwsum, bwsum);

  finalize_kernel<<<128, 256, 0, stream>>>(x, fwsum, bwsum, out);
}
